// AttentionModule_26525718020806
// MI455X (gfx1250) — matrix operand order repaired and checked by reading, not compiled, not run on hardware
//
#include <hip/hip_runtime.h>
#include <cmath>

typedef _Float16 half8_t __attribute__((ext_vector_type(8)));
typedef _Float16 h16_t   __attribute__((ext_vector_type(16)));
typedef float    f8_t    __attribute__((ext_vector_type(8)));

namespace {
constexpr int kL  = 12;    // layers
constexpr int kD  = 256;   // d_model
constexpr int kNH = 4;     // heads
constexpr int kDH = 64;    // head dim
constexpr int kN  = 2048;  // sequence points
constexpr int kB  = 2;     // batch
constexpr int kTB = 4;     // (descriptor t, batch b) pairs

enum GemmMode { MODE_PLAIN = 0, MODE_QK = 1, MODE_V = 2, MODE_BN = 3, MODE_RESID = 4 };
}

// ---- WMMA fragment builders (f16 16x16x32, wave32; contiguous k order, same in A and B) ----
// A (16xK=32): lane l (row M = l&15, khalf = l>>4) holds K = khalf*16 + [0..16)
//              -> one contiguous 16-half run (two 16B loads).
__device__ __forceinline__ h16_t afrag(const _Float16* __restrict__ row, int k0, int khalf) {
  half8_t a0 = *(const half8_t*)(row + k0 + khalf * 16);
  half8_t a1 = *(const half8_t*)(row + k0 + khalf * 16 + 8);
  h16_t A;
#pragma unroll
  for (int i = 0; i < 8; ++i) { A[i] = a0[i]; A[i + 8] = a1[i]; }
  return A;
}
// B (K=32 x 16): lane l (col N = l&15, khalf = l>>4) holds K = khalf*16 + [0..16)
//                -> one contiguous 16-half (32B) run.
__device__ __forceinline__ h16_t bfrag(const _Float16* __restrict__ p) {
  half8_t b0 = *(const half8_t*)(p);
  half8_t b1 = *(const half8_t*)(p + 8);
  h16_t B;
#pragma unroll
  for (int i = 0; i < 8; ++i) { B[i] = b0[i]; B[i + 8] = b1[i]; }
  return B;
}

// ---------------------------------------------------------------------------
__global__ __launch_bounds__(256) void k_f32_to_f16(const float* __restrict__ s,
                                                    _Float16* __restrict__ d, int n) {
  int i = blockIdx.x * 256 + threadIdx.x;
  if (i < n) d[i] = (_Float16)s[i];
}

__global__ __launch_bounds__(256) void k_init_out(const float* __restrict__ d0,
                                                  const float* __restrict__ d1,
                                                  float* __restrict__ out, int n) {
  int i = blockIdx.x * 256 + threadIdx.x;
  if (i < n) { out[i] = d0[i]; out[n + i] = d1[i]; }
}

// master fp32 desc [tb][c][n] -> f16 activation [tb][n][c], LDS-tiled 64x64
// transpose: coalesced 256B row loads, coalesced 128B column stores.
__global__ __launch_bounds__(256) void k_desc_to_half(const float* __restrict__ dout,
                                                      _Float16* __restrict__ xh) {
  __shared__ _Float16 tile[64][66];           // 132B pitch = 33 banks, conflict-free
  const int tb = blockIdx.y;
  const int ct = blockIdx.x / (kN / 64);      // 0..3   (channel tile)
  const int nt = blockIdx.x % (kN / 64);      // 0..31  (point tile)
  const int c0 = ct * 64, n0 = nt * 64;

  const int nl = threadIdx.x & 63;            // lane along n (coalesced loads)
  const int cq = threadIdx.x >> 6;            // 0..3
  const float* src = dout + ((size_t)tb * kD + c0) * kN + n0;
#pragma unroll
  for (int i = 0; i < 16; ++i) {
    const int cl = cq * 16 + i;
    tile[cl][nl] = (_Float16)src[(size_t)cl * kN + nl];
  }
  __syncthreads();
  const int co = threadIdx.x & 63;            // lane along c (coalesced stores)
  const int nq = threadIdx.x >> 6;
  _Float16* dst = xh + ((size_t)tb * kN + n0) * kD + c0;
#pragma unroll
  for (int i = 0; i < 16; ++i) {
    const int nl2 = nq * 16 + i;
    dst[(size_t)nl2 * kD + co] = tile[co][nl2];
  }
}

// ---------------------------------------------------------------------------
// C[o,n] = sum_k W[o,k] * act[n,k] (+bias, fused epilogue). One wave = 32x64
// output tile (2 o-tiles x 4 n-tiles, 8 accumulators): every B fragment is
// shared by two A fragments -> 12 b128 loads per 8 WMMAs. Shapes compile-time.
template <int O, int I, int cA, int MODE>
__global__ __launch_bounds__(256) void k_wmma_gemm(
    const _Float16* __restrict__ W,      // [O][I] f16
    const _Float16* __restrict__ actA,   // [kTB][kN][cA] f16
    const _Float16* __restrict__ actB2,  // [kTB][kN][I-cA] f16 (only if cA<I)
    const float* __restrict__ bias,      // [O]
    _Float16* __restrict__ outh,
    float* __restrict__ resid,           // d_out for MODE_RESID
    const float* __restrict__ gamma, const float* __restrict__ beta) {
  constexpr int cB = I - cA;
  const int lane  = threadIdx.x & 31;
  const int wv    = threadIdx.x >> 5;
  const int tb    = blockIdx.y;
  const int khalf = lane >> 4;
  const int l15   = lane & 15;
  const int tile  = blockIdx.x * 8 + wv;
  constexpr int nt4 = kN / 64;
  const int o_base = (tile / nt4) * 32;
  const int n_base = (tile % nt4) * 64;

  const _Float16* wrow0 = W + (size_t)(o_base + l15) * I;
  const _Float16* wrow1 = W + (size_t)(o_base + 16 + l15) * I;
  const int boff = khalf * 16;

  f8_t acc[2][4];
#pragma unroll
  for (int oi = 0; oi < 2; ++oi)
#pragma unroll
    for (int j = 0; j < 4; ++j)
#pragma unroll
      for (int r = 0; r < 8; ++r) acc[oi][j][r] = 0.0f;

  {
    const _Float16* browA[4];
#pragma unroll
    for (int j = 0; j < 4; ++j)
      browA[j] = actA + ((size_t)tb * kN + n_base + j * 16 + l15) * cA + boff;
#pragma unroll 2
    for (int k0 = 0; k0 < cA; k0 += 32) {
      h16_t A0 = afrag(wrow0, k0, khalf);
      h16_t A1 = afrag(wrow1, k0, khalf);
      __builtin_prefetch(wrow0 + k0 + 32, 0, 1);   // global_prefetch_b8
#pragma unroll
      for (int j = 0; j < 4; ++j) {
        h16_t Bf = bfrag(browA[j] + k0);
        acc[0][j] = __builtin_amdgcn_wmma_f32_16x16x32_f16(false, A0, false, Bf,
                                                           (short)0, acc[0][j], false, false);
        acc[1][j] = __builtin_amdgcn_wmma_f32_16x16x32_f16(false, A1, false, Bf,
                                                           (short)0, acc[1][j], false, false);
      }
    }
  }
  if constexpr (cB > 0) {
    const _Float16* browB[4];
#pragma unroll
    for (int j = 0; j < 4; ++j)
      browB[j] = actB2 + ((size_t)tb * kN + n_base + j * 16 + l15) * cB + boff;
#pragma unroll 2
    for (int k0 = 0; k0 < cB; k0 += 32) {
      h16_t A0 = afrag(wrow0 + cA, k0, khalf);
      h16_t A1 = afrag(wrow1 + cA, k0, khalf);
      __builtin_prefetch(wrow0 + cA + k0 + 32, 0, 1);
#pragma unroll
      for (int j = 0; j < 4; ++j) {
        h16_t Bf = bfrag(browB[j] + k0);
        acc[0][j] = __builtin_amdgcn_wmma_f32_16x16x32_f16(false, A0, false, Bf,
                                                           (short)0, acc[0][j], false, false);
        acc[1][j] = __builtin_amdgcn_wmma_f32_16x16x32_f16(false, A1, false, Bf,
                                                           (short)0, acc[1][j], false, false);
      }
    }
  }

#pragma unroll
  for (int oi = 0; oi < 2; ++oi) {
#pragma unroll
    for (int j = 0; j < 4; ++j) {
      const int nc = n_base + j * 16 + l15;
#pragma unroll
      for (int r = 0; r < 8; ++r) {
        const int o = o_base + oi * 16 + (khalf << 3) + r;   // C-fragment rows
        float v = acc[oi][j][r] + bias[o];
        if constexpr (MODE == MODE_PLAIN) {
          outh[((size_t)tb * kN + nc) * O + o] = (_Float16)v;
        } else if constexpr (MODE == MODE_QK) {    // head-split [h][n][idx]
          const int hh = o & 3, idx = o >> 2;      // d = idx*NH + h (reshape)
          outh[(((size_t)tb * kNH + hh) * kN + nc) * kDH + idx] = (_Float16)v;
        } else if constexpr (MODE == MODE_V) {     // transposed [h][idx][n]
          const int hh = o & 3, idx = o >> 2;
          outh[(((size_t)tb * kNH + hh) * kDH + idx) * kN + nc] = (_Float16)v;
        } else if constexpr (MODE == MODE_BN) {    // BN(inference) + ReLU
          const float invs = rsqrtf(1.0f + 1e-5f);
          float hv = gamma[o] * (v * invs) + beta[o];
          outh[((size_t)tb * kN + nc) * O + o] = (_Float16)fmaxf(hv, 0.0f);
        } else {                                   // residual into fp32 master
          resid[(size_t)tb * kD * kN + (size_t)o * kN + nc] += v;
        }
      }
    }
  }
}

// ---------------------------------------------------------------------------
// Flash-style attention: one wave per (16-query tile, head, tb). Streams key
// chunks of 64 (4 score tiles, 2 P fragments): the per-row cross-lane softmax
// reductions amortize over twice the keys. 1/sqrt(dh)=0.125 is folded into the
// Q fragment (exact power-of-2 scale in f16).
__global__ __launch_bounds__(256) void k_attn(const _Float16* __restrict__ qh,
                                              const _Float16* __restrict__ kh,
                                              const _Float16* __restrict__ vt,
                                              _Float16* __restrict__ msgh, int cross) {
  __shared__ __attribute__((aligned(16))) _Float16 plds[8][16][72];  // 144B rows
  const int lane  = threadIdx.x & 31;
  const int wv    = threadIdx.x >> 5;
  const int khalf = lane >> 4;
  const int l15   = lane & 15;
  const int gw    = blockIdx.x * 8 + wv;
  const int h     = gw & (kNH - 1);
  const int qtile = gw >> 2;
  const int tb    = blockIdx.y;
  const int t = tb >> 1, bb = tb & 1;
  const int sb = (cross ? (t ^ 1) : t) * 2 + bb;   // source descriptor for K/V

  const _Float16* qrow  = qh + (((size_t)tb * kNH + h) * kN + qtile * 16 + l15) * kDH;
  const _Float16* kbase = kh + ((size_t)sb * kNH + h) * kN * kDH;
  const _Float16* vbase = vt + ((size_t)sb * kNH + h) * kDH * kN;

  h16_t Aq0 = afrag(qrow, 0, khalf);    // Q rows fixed for whole wave
  h16_t Aq1 = afrag(qrow, 32, khalf);
#pragma unroll
  for (int i = 0; i < 16; ++i) {        // fold 1/sqrt(64): exact in f16
    Aq0[i] = Aq0[i] * (_Float16)0.125f;
    Aq1[i] = Aq1[i] * (_Float16)0.125f;
  }

  f8_t Oacc[4];
#pragma unroll
  for (int dt = 0; dt < 4; ++dt)
#pragma unroll
    for (int r = 0; r < 8; ++r) Oacc[dt][r] = 0.0f;
  float rmax[8], rsum[8];
#pragma unroll
  for (int r = 0; r < 8; ++r) { rmax[r] = -1e30f; rsum[r] = 0.0f; }

  for (int m0 = 0; m0 < kN; m0 += 64) {
    f8_t s[4];
#pragma unroll
    for (int j = 0; j < 4; ++j)
#pragma unroll
      for (int r = 0; r < 8; ++r) s[j][r] = 0.0f;
#pragma unroll
    for (int j = 0; j < 4; ++j) {
      const _Float16* krow = kbase + (size_t)(m0 + j * 16 + l15) * kDH;
      s[j] = __builtin_amdgcn_wmma_f32_16x16x32_f16(false, Aq0, false,
                bfrag(krow + khalf * 16), (short)0, s[j], false, false);
      s[j] = __builtin_amdgcn_wmma_f32_16x16x32_f16(false, Aq1, false,
                bfrag(krow + 32 + khalf * 16), (short)0, s[j], false, false);
    }
    // online softmax per C-fragment row; reductions stay within 16-lane halves
#pragma unroll
    for (int r = 0; r < 8; ++r) {
      const float a0 = s[0][r], a1 = s[1][r], a2 = s[2][r], a3 = s[3][r];
      float mx = fmaxf(fmaxf(a0, a1), fmaxf(a2, a3));
#pragma unroll
      for (int off = 1; off < 16; off <<= 1) mx = fmaxf(mx, __shfl_xor(mx, off, 32));
      const float nm   = fmaxf(rmax[r], mx);
      const float corr = __expf(rmax[r] - nm);
      const float p0 = __expf(a0 - nm), p1 = __expf(a1 - nm);
      const float p2 = __expf(a2 - nm), p3 = __expf(a3 - nm);
      float ps = (p0 + p1) + (p2 + p3);
#pragma unroll
      for (int off = 1; off < 16; off <<= 1) ps += __shfl_xor(ps, off, 32);
      rsum[r] = rsum[r] * corr + ps;
      rmax[r] = nm;
#pragma unroll
      for (int dt = 0; dt < 4; ++dt) Oacc[dt][r] *= corr;
      const int nloc = (khalf << 3) + r;
      plds[wv][nloc][l15]      = (_Float16)p0;
      plds[wv][nloc][16 + l15] = (_Float16)p1;
      plds[wv][nloc][32 + l15] = (_Float16)p2;
      plds[wv][nloc][48 + l15] = (_Float16)p3;
    }
    asm volatile("s_wait_dscnt 0x0" ::: "memory");  // LDS RAW: P staged -> A frags
    h16_t Ap0, Ap1;
    {
      const half8_t pa = *(const half8_t*)&plds[wv][l15][khalf * 16];
      const half8_t pb = *(const half8_t*)&plds[wv][l15][khalf * 16 + 8];
      const half8_t pc = *(const half8_t*)&plds[wv][l15][32 + khalf * 16];
      const half8_t pd = *(const half8_t*)&plds[wv][l15][40 + khalf * 16];
#pragma unroll
      for (int i2 = 0; i2 < 8; ++i2) {
        Ap0[i2] = pa[i2]; Ap0[i2 + 8] = pb[i2];
        Ap1[i2] = pc[i2]; Ap1[i2 + 8] = pd[i2];
      }
    }
#pragma unroll
    for (int dt = 0; dt < 4; ++dt) {
      const _Float16* vrow = vbase + (size_t)(dt * 16 + l15) * kN + m0 + khalf * 16;
      Oacc[dt] = __builtin_amdgcn_wmma_f32_16x16x32_f16(false, Ap0, false,
                    bfrag(vrow), (short)0, Oacc[dt], false, false);
      Oacc[dt] = __builtin_amdgcn_wmma_f32_16x16x32_f16(false, Ap1, false,
                    bfrag(vrow + 32), (short)0, Oacc[dt], false, false);
    }
  }

  float inv[8];
#pragma unroll
  for (int r = 0; r < 8; ++r) inv[r] = 1.0f / rsum[r];   // 8 divides, not 32
#pragma unroll
  for (int dt = 0; dt < 4; ++dt) {
#pragma unroll
    for (int r = 0; r < 8; ++r) {
      const int n   = qtile * 16 + (khalf << 3) + r;
      const int idx = dt * 16 + l15;
      // msg channel d = idx*NH + h, layout [tb][n][256]
      msgh[((size_t)tb * kN + n) * kD + idx * kNH + h] = (_Float16)(Oacc[dt][r] * inv[r]);
    }
  }
}

// ---------------------------------------------------------------------------
extern "C" void kernel_launch(void* const* d_in, const int* in_sizes, int n_in,
                              void* d_out, int out_size, void* d_ws, size_t ws_size,
                              hipStream_t stream) {
  (void)in_sizes; (void)n_in; (void)out_size; (void)ws_size;
  const float* desc0 = (const float*)d_in[0];
  const float* desc1 = (const float*)d_in[1];
  const float* Wq = (const float*)d_in[2];   const float* bq = (const float*)d_in[3];
  const float* Wk = (const float*)d_in[4];   const float* bk = (const float*)d_in[5];
  const float* Wv = (const float*)d_in[6];   const float* bv = (const float*)d_in[7];
  const float* Wm = (const float*)d_in[8];   const float* bm = (const float*)d_in[9];
  const float* W1 = (const float*)d_in[10];  const float* b1 = (const float*)d_in[11];
  const float* gamma = (const float*)d_in[12];
  const float* beta  = (const float*)d_in[13];
  const float* W2 = (const float*)d_in[14];  const float* b2 = (const float*)d_in[15];
  float* dout = (float*)d_out;

  // ---- workspace layout (all f16; sizes are multiples of 256B) ----
  char* p = (char*)d_ws;
  auto take = [&](size_t elems) { _Float16* q = (_Float16*)p; p += elems * sizeof(_Float16); return q; };
  const size_t eW   = (size_t)kL * kD * kD;            // 786432
  const size_t eW1  = (size_t)kL * 2 * kD * 2 * kD;    // 3145728
  const size_t eW2  = (size_t)kL * kD * 2 * kD;        // 1572864
  const size_t eAct = (size_t)kTB * kN * kD;           // 2097152
  const size_t eH   = (size_t)kTB * kN * 2 * kD;       // 4194304
  _Float16* WqH = take(eW);   _Float16* WkH = take(eW);
  _Float16* WvH = take(eW);   _Float16* WmH = take(eW);
  _Float16* W1H = take(eW1);  _Float16* W2H = take(eW2);
  _Float16* xh   = take(eAct);   // current descriptors  [tb][n][256]
  _Float16* qhB  = take(eAct);   // Q  [tb][h][n][64]
  _Float16* khB  = take(eAct);   // K  [tb][h][m][64]
  _Float16* vtB  = take(eAct);   // V^T [tb][h][64][m]
  _Float16* msgh = take(eAct);   // attention output [tb][n][256]
  _Float16* m2h  = take(eAct);   // after Wm conv     [tb][n][256]
  _Float16* hh   = take(eH);     // after W1+BN+ReLU  [tb][n][512]

  const dim3 blk(256);
  // per-call: weights -> f16, residual master = inputs (graph-replay safe)
  k_f32_to_f16<<<(int)(eW  / 256), blk, 0, stream>>>(Wq, WqH, (int)eW);
  k_f32_to_f16<<<(int)(eW  / 256), blk, 0, stream>>>(Wk, WkH, (int)eW);
  k_f32_to_f16<<<(int)(eW  / 256), blk, 0, stream>>>(Wv, WvH, (int)eW);
  k_f32_to_f16<<<(int)(eW  / 256), blk, 0, stream>>>(Wm, WmH, (int)eW);
  k_f32_to_f16<<<(int)(eW1 / 256), blk, 0, stream>>>(W1, W1H, (int)eW1);
  k_f32_to_f16<<<(int)(eW2 / 256), blk, 0, stream>>>(W2, W2H, (int)eW2);
  k_init_out<<<(kB * kD * kN) / 256, blk, 0, stream>>>(desc0, desc1, dout, kB * kD * kN);

  const dim3 gT(128, kTB);     // transpose: (4 c-tiles * 32 n-tiles)
  const dim3 g256(32, kTB);    // O=256: (8 o2-tiles * 32 n4-tiles)/8 waves
  const dim3 g512(64, kTB);    // O=512: (16 * 32)/8
  const dim3 gAtt(64, kTB);    // (128 q-tiles * 4 heads)/8 waves

  for (int i = 0; i < kL; ++i) {
    const _Float16* WqL = WqH + (size_t)i * kD * kD;
    const _Float16* WkL = WkH + (size_t)i * kD * kD;
    const _Float16* WvL = WvH + (size_t)i * kD * kD;
    const _Float16* WmL = WmH + (size_t)i * kD * kD;
    const _Float16* W1L = W1H + (size_t)i * 2 * kD * 2 * kD;
    const _Float16* W2L = W2H + (size_t)i * kD * 2 * kD;

    k_desc_to_half<<<gT, blk, 0, stream>>>(dout, xh);

    k_wmma_gemm<kD, kD, kD, MODE_QK><<<g256, blk, 0, stream>>>(
        WqL, xh, nullptr, bq + i * kD, qhB, nullptr, nullptr, nullptr);
    k_wmma_gemm<kD, kD, kD, MODE_QK><<<g256, blk, 0, stream>>>(
        WkL, xh, nullptr, bk + i * kD, khB, nullptr, nullptr, nullptr);
    k_wmma_gemm<kD, kD, kD, MODE_V><<<g256, blk, 0, stream>>>(
        WvL, xh, nullptr, bv + i * kD, vtB, nullptr, nullptr, nullptr);

    k_attn<<<gAtt, blk, 0, stream>>>(qhB, khB, vtB, msgh, i & 1);

    k_wmma_gemm<kD, kD, kD, MODE_PLAIN><<<g256, blk, 0, stream>>>(
        WmL, msgh, nullptr, bm + i * kD, m2h, nullptr, nullptr, nullptr);
    // W1 over concat([x, msg]) with fused BN(inference) + ReLU
    k_wmma_gemm<2 * kD, 2 * kD, kD, MODE_BN><<<g512, blk, 0, stream>>>(
        W1L, xh, m2h, b1 + i * 2 * kD, hh, nullptr, gamma + i * 2 * kD, beta + i * 2 * kD);
    // W2 + residual accumulate into fp32 master (d_out)
    k_wmma_gemm<kD, 2 * kD, 2 * kD, MODE_RESID><<<g256, blk, 0, stream>>>(
        W2L, hh, nullptr, b2 + i * kD, nullptr, dout, nullptr, nullptr);
  }
}
